// InvariantPointAttention_frames_70841190580552
// MI455X (gfx1250) — matrix operand order repaired and checked by reading, not compiled, not run on hardware
//
#include <hip/hip_runtime.h>
#include <hip/hip_bf16.h>
#include <math.h>

#define B_    2
#define N_    1024
#define BN_   (B_ * N_)
#define CS_   384
#define CZ_   128
#define CH_   16
#define H_    12
#define PQ_   4
#define PV_   8
#define K_    50
#define HALF_ 512
#define CAT_  2112   // H*(CZ + CH + PV*4)

typedef __attribute__((ext_vector_type(16))) _Float16 v16h;
typedef __attribute__((ext_vector_type(8)))  _Float16 v8h;
typedef __attribute__((ext_vector_type(8)))  float    v8f;
typedef __attribute__((ext_vector_type(4)))  float    f32x4;

// ---------------------------------------------------------------------------
// WMMA fragment loaders.
// A fragment (16-bit A 16x32): lane<16 holds K {kb+0..7, kb+16..23},
// lane>=16 holds K {kb+8..15, kb+24..31}; M = lane%16.  aoff = lane<16?0:8.
// B fragment: lane holds col N = lane%16, same K order as A (boff/2 = aoff).
// All offsets are multiples of 8 f32 (32B) / 16 f32 (64B) -> b128 loads legal.
// ---------------------------------------------------------------------------
__device__ __forceinline__ v16h load_a_f32(const float* __restrict__ p, int kb, int aoff) {
    f32x4 a = *(const f32x4*)(p + kb + aoff);
    f32x4 b = *(const f32x4*)(p + kb + aoff + 4);
    f32x4 c = *(const f32x4*)(p + kb + 16 + aoff);
    f32x4 d = *(const f32x4*)(p + kb + 16 + aoff + 4);
    v16h A;
#pragma unroll
    for (int i = 0; i < 4; ++i) {
        A[i]      = (_Float16)a[i];
        A[i + 4]  = (_Float16)b[i];
        A[i + 8]  = (_Float16)c[i];
        A[i + 12] = (_Float16)d[i];
    }
    return A;
}

__device__ __forceinline__ v16h load_a_f16(const _Float16* __restrict__ p, int kb, int aoff) {
    v8h lo = *(const v8h*)(p + kb + aoff);
    v8h hi = *(const v8h*)(p + kb + 16 + aoff);
    v16h A;
#pragma unroll
    for (int i = 0; i < 8; ++i) { A[i] = lo[i]; A[i + 8] = hi[i]; }
    return A;
}

__device__ __forceinline__ v16h load_b_f32(const float* __restrict__ wrow, int kb, int boff) {
    f32x4 x0 = *(const f32x4*)(wrow + kb + boff / 2);
    f32x4 x1 = *(const f32x4*)(wrow + kb + boff / 2 + 4);
    f32x4 x2 = *(const f32x4*)(wrow + kb + 16 + boff / 2);
    f32x4 x3 = *(const f32x4*)(wrow + kb + 16 + boff / 2 + 4);
    v16h Bv;
#pragma unroll
    for (int i = 0; i < 4; ++i) {
        Bv[i]      = (_Float16)x0[i];
        Bv[i + 4]  = (_Float16)x1[i];
        Bv[i + 8]  = (_Float16)x2[i];
        Bv[i + 12] = (_Float16)x3[i];
    }
    return Bv;
}

// ---------------------------------------------------------------------------
// Kernel 1: fused projection GEMM  (2048 x 384) @ (384 x 1152)
// cols: [0,192)=q  [192,576)=kv  [576,720)=qp  [720,1152)=kvp
// One wave computes a 32x64 output block (2 M-tiles x 4 N-tiles, 8 accums):
// A loaded once per M-tile per K-step, 8 WMMAs issued back-to-back.
// ---------------------------------------------------------------------------
__global__ __launch_bounds__(32) void k_proj(
    const float* __restrict__ s,
    const float* __restrict__ Wq,  const float* __restrict__ bq,
    const float* __restrict__ Wkv, const float* __restrict__ bkv,
    const float* __restrict__ Wqp, const float* __restrict__ bqp,
    const float* __restrict__ Wkvp,const float* __restrict__ bkvp,
    float* __restrict__ qb,  float* __restrict__ kvb,
    float* __restrict__ qpb, float* __restrict__ kvpb)
{
    const int NG = 18;                       // 72 N-tiles / 4
    int mg = blockIdx.x / NG;                // 64 M-groups (32 rows each)
    int ng = blockIdx.x % NG;

    int lane = threadIdx.x;
    int l16  = lane & 15;
    int aoff = (lane < 16) ? 0 : 8;
    int boff = (lane < 16) ? 0 : 16;

    // resolve the 4 N-subtiles (segment boundary at tile 45 is not 4-aligned,
    // so each subtile carries its own weight/bias/output pointers)
    const float* wrow[4]; const float* bias[4]; float* outp[4];
    int ncols[4]; int col[4];
#pragma unroll
    for (int j = 0; j < 4; ++j) {
        int nt = ng * 4 + j;
        const float* W; int cb;
        if (nt < 12)      { W = Wq;   bias[j] = bq;   outp[j] = qb;   ncols[j] = 192; cb = nt * 16; }
        else if (nt < 36) { W = Wkv;  bias[j] = bkv;  outp[j] = kvb;  ncols[j] = 384; cb = (nt - 12) * 16; }
        else if (nt < 45) { W = Wqp;  bias[j] = bqp;  outp[j] = qpb;  ncols[j] = 144; cb = (nt - 36) * 16; }
        else              { W = Wkvp; bias[j] = bkvp; outp[j] = kvpb; ncols[j] = 432; cb = (nt - 45) * 16; }
        col[j]  = cb + l16;
        wrow[j] = W + (size_t)col[j] * CS_;
    }

    const float* arow0 = s + (size_t)(mg * 32 + l16) * CS_;
    const float* arow1 = arow0 + (size_t)16 * CS_;

    v8f acc0[4] = {}; v8f acc1[4] = {};
    for (int kb = 0; kb < CS_; kb += 32) {
        v16h A0 = load_a_f32(arow0, kb, aoff);
        v16h A1 = load_a_f32(arow1, kb, aoff);
#pragma unroll
        for (int j = 0; j < 4; ++j) {
            v16h Bm = load_b_f32(wrow[j], kb, boff);
            acc0[j] = __builtin_amdgcn_wmma_f32_16x16x32_f16(false, A0, false, Bm,
                                                             (short)0, acc0[j], false, false);
            acc1[j] = __builtin_amdgcn_wmma_f32_16x16x32_f16(false, A1, false, Bm,
                                                             (short)0, acc1[j], false, false);
        }
    }

    int mb0 = mg * 32 + ((lane < 16) ? 0 : 8);   // C/D: VGPR r -> M=r+(lane<16?0:8)
#pragma unroll
    for (int j = 0; j < 4; ++j) {
        float bv = bias[j][col[j]];
#pragma unroll
        for (int r = 0; r < 8; ++r) {
            outp[j][(size_t)(mb0 + r) * ncols[j] + col[j]]      = acc0[j][r] + bv;
            outp[j][(size_t)(mb0 + 16 + r) * ncols[j] + col[j]] = acc1[j][r] + bv;
        }
    }
}

// ---------------------------------------------------------------------------
// Kernel 2: rigid transforms -> q_pts (B,N,H*PQ,3), k_pts, v_pts
// projection layout is component-major: x[i*HP + j] = comp i of point j.
// ---------------------------------------------------------------------------
__global__ __launch_bounds__(256) void k_points(
    const float* __restrict__ qpb, const float* __restrict__ kvpb,
    const float* __restrict__ rot, const float* __restrict__ trans,
    float* __restrict__ q_pts, float* __restrict__ k_pts, float* __restrict__ v_pts)
{
    int t = blockIdx.x * blockDim.x + threadIdx.x;
    if (t >= BN_ * 192) return;
    int bn = t / 192, j = t % 192;
    const float* R = rot   + (size_t)bn * 9;
    const float* T = trans + (size_t)bn * 3;

    if (j < 48) {                                   // q_pts: HP = 48
        const float* src = qpb + (size_t)bn * 144;
        float x0 = src[j], x1 = src[48 + j], x2 = src[96 + j];
        float* dst = q_pts + (size_t)bn * 144 + (size_t)j * 3;
        dst[0] = R[0]*x0 + R[1]*x1 + R[2]*x2 + T[0];
        dst[1] = R[3]*x0 + R[4]*x1 + R[5]*x2 + T[1];
        dst[2] = R[6]*x0 + R[7]*x1 + R[8]*x2 + T[2];
    } else {                                        // kv_pts: HP = 144
        int j2 = j - 48;
        const float* src = kvpb + (size_t)bn * 432;
        float x0 = src[j2], x1 = src[144 + j2], x2 = src[288 + j2];
        float o0 = R[0]*x0 + R[1]*x1 + R[2]*x2 + T[0];
        float o1 = R[3]*x0 + R[4]*x1 + R[5]*x2 + T[1];
        float o2 = R[6]*x0 + R[7]*x1 + R[8]*x2 + T[2];
        int h = j2 / 12, p = j2 % 12;
        float* dst = (p < PQ_)
            ? k_pts + (size_t)bn * 144 + (size_t)(h * PQ_ + p) * 3
            : v_pts + (size_t)bn * 288 + (size_t)(h * PV_ + (p - PQ_)) * 3;
        dst[0] = o0; dst[1] = o1; dst[2] = o2;
    }
}

// ---------------------------------------------------------------------------
// Kernel 3: top-K=50 nearest neighbors per residue (iterative argmin in LDS,
// lower-index wins on ties to match jax.lax.top_k).
// ---------------------------------------------------------------------------
__global__ __launch_bounds__(256) void k_topk(
    const float* __restrict__ trans, int* __restrict__ nidx)
{
    __shared__ float d2s[N_];
    __shared__ float rv[256];
    __shared__ int   ri[256];

    int bn = blockIdx.x;
    int b  = bn >> 10;
    int tid = threadIdx.x;
    float tx = trans[(size_t)bn*3], ty = trans[(size_t)bn*3+1], tz = trans[(size_t)bn*3+2];
    const float* tb = trans + (size_t)b * N_ * 3;

    for (int m = tid; m < N_; m += 256) {
        float dx = tb[m*3] - tx, dy = tb[m*3+1] - ty, dz = tb[m*3+2] - tz;
        d2s[m] = dx*dx + dy*dy + dz*dz;
    }
    __syncthreads();

    for (int it = 0; it < K_; ++it) {
        float best = 3.4e38f; int bi = N_;
        for (int m = tid; m < N_; m += 256) {
            float v = d2s[m];
            if (v < best) { best = v; bi = m; }
        }
        rv[tid] = best; ri[tid] = bi;
        __syncthreads();
        for (int s2 = 128; s2 > 0; s2 >>= 1) {
            if (tid < s2) {
                float v2 = rv[tid + s2]; int i2 = ri[tid + s2];
                if (v2 < rv[tid] || (v2 == rv[tid] && i2 < ri[tid])) {
                    rv[tid] = v2; ri[tid] = i2;
                }
            }
            __syncthreads();
        }
        if (tid == 0) { nidx[(size_t)bn * K_ + it] = ri[0]; d2s[ri[0]] = 3.4e38f; }
        __syncthreads();
    }
}

// ---------------------------------------------------------------------------
// Kernel 4: gathered attention per query (one 256-thread workgroup per (b,n)).
// zn (used twice) staged in LDS via b128 loads; kn/vn/k_pts/v_pts streamed
// from L2 since every gathered element is consumed exactly once. Writes the
// 2112-wide concat row directly in f16 (WMMA-A-ready for the output GEMM).
// ---------------------------------------------------------------------------
__global__ __launch_bounds__(256) void k_attn(
    const float* __restrict__ qb,    const float* __restrict__ kvb,
    const float* __restrict__ q_pts, const float* __restrict__ k_pts,
    const float* __restrict__ v_pts, const float* __restrict__ z,
    const float* __restrict__ rot,   const float* __restrict__ trans,
    const float* __restrict__ mask,  const int* __restrict__ nidx,
    const float* __restrict__ Wb,    const float* __restrict__ bb,
    const float* __restrict__ hwraw, _Float16* __restrict__ cat)
{
    __shared__ __align__(16) float zn_s[K_ * CZ_];   // 25.6 KB
    __shared__ float q_s[192];
    __shared__ float qp_s[144];
    __shared__ float bp_s[K_ * H_];
    __shared__ float a_s[H_ * 64];     // padded rows of 64
    __shared__ float opt_s[288];
    __shared__ float mk_s[K_];
    __shared__ int   ni_s[K_];
    __shared__ float hw_s[H_];
    __shared__ float R_s[9];
    __shared__ float T_s[3];

    int bn = blockIdx.x, b = bn >> 10, n = bn & (N_ - 1);
    int tid = threadIdx.x;

    if (tid < K_) ni_s[tid] = nidx[(size_t)bn * K_ + tid];
    __syncthreads();

    if (tid < 192)                 q_s[tid]        = qb[(size_t)bn * 192 + tid];
    if (tid >= 192 && tid < 336)   qp_s[tid - 192] = q_pts[(size_t)bn * 144 + (tid - 192)];
    if (tid < H_) hw_s[tid] = log1pf(expf(hwraw[tid])) * 0.13608276348795434f; // sqrt(1/54)
    if (tid < 9)  R_s[tid] = rot[(size_t)bn * 9 + tid];
    if (tid < 3)  T_s[tid] = trans[(size_t)bn * 3 + tid];
    if (tid < K_) {
        mk_s[tid] = 100000.0f * (mask[b * N_ + n] * mask[b * N_ + ni_s[tid]] - 1.0f);
        __builtin_prefetch(kvb + (size_t)(b * N_ + ni_s[tid]) * 384, 0, 1);
    }
    // zn tile (50 x 128 f32) -> LDS, 16B granularity
    for (int idx = tid; idx < K_ * (CZ_ / 4); idx += 256) {
        int k = idx >> 5, c4 = idx & 31;
        const f32x4* src = (const f32x4*)(z + (((size_t)b * HALF_ + (n & (HALF_-1))) * HALF_
                                               + (ni_s[k] & (HALF_-1))) * CZ_);
        *((f32x4*)(zn_s + k * CZ_) + c4) = src[c4];
    }
    __syncthreads();

    // pair bias: b_pair[k][h] = zn[k] . Wb[h] + bb[h]
    for (int t = tid; t < K_ * H_; t += 256) {
        int k = t / H_, h = t - k * H_;
        const f32x4* zr = (const f32x4*)(zn_s + k * CZ_);
        const f32x4* wr = (const f32x4*)(Wb + h * CZ_);
        float acc = bb[h];
        for (int c = 0; c < CZ_ / 4; ++c) {
            f32x4 zv = zr[c], wv = wr[c];
            acc += zv[0]*wv[0] + zv[1]*wv[1] + zv[2]*wv[2] + zv[3]*wv[3];
        }
        bp_s[t] = acc;
    }
    __syncthreads();

    // logits
    for (int t = tid; t < H_ * K_; t += 256) {
        int h = t / K_, k = t - h * K_;
        int nk = ni_s[k];
        const float* kvr = kvb + (size_t)(b * N_ + nk) * 384 + h * 32;
        float qd = 0.f;
        for (int c = 0; c < CH_; ++c) qd += q_s[h * CH_ + c] * kvr[c];
        const float* kpr = k_pts + (size_t)(b * N_ + nk) * 144 + h * 12;
        float d2 = 0.f;
        for (int pi = 0; pi < 12; ++pi) { float d = qp_s[h * 12 + pi] - kpr[pi]; d2 += d * d; }
        a_s[h * 64 + k] = qd * 0.14433756729740643f          // sqrt(1/48)
                        + 0.5773502691896258f * bp_s[k * H_ + h]
                        - 0.5f * hw_s[h] * d2 + mk_s[k];
    }
    __syncthreads();

    // softmax over K per head
    if (tid < H_) {
        int h = tid;
        float m = -3.4e38f;
        for (int k = 0; k < K_; ++k) m = fmaxf(m, a_s[h * 64 + k]);
        float sum = 0.f;
        for (int k = 0; k < K_; ++k) { float e = expf(a_s[h * 64 + k] - m); a_s[h * 64 + k] = e; sum += e; }
        float inv = 1.0f / sum;
        for (int k = 0; k < K_; ++k) a_s[h * 64 + k] *= inv;
    }
    __syncthreads();

    // o = a @ vn   -> cat[0:192)
    for (int t = tid; t < 192; t += 256) {
        int h = t >> 4, c = t & 15;
        float acc = 0.f;
        for (int k = 0; k < K_; ++k)
            acc += a_s[h * 64 + k] * kvb[(size_t)(b * N_ + ni_s[k]) * 384 + h * 32 + 16 + c];
        cat[(size_t)bn * CAT_ + t] = (_Float16)acc;
    }
    // o_pt (global frame) = a @ vpn
    for (int t = tid; t < 288; t += 256) {
        int h = t / 24;
        float acc = 0.f;
        for (int k = 0; k < K_; ++k)
            acc += a_s[h * 64 + k] * v_pts[(size_t)(b * N_ + ni_s[k]) * 288 + t];
        opt_s[t] = acc;
    }
    __syncthreads();
    // inverse transform + norm -> cat[192:576)
    for (int t = tid; t < 96; t += 256) {
        float u0 = opt_s[t * 3]     - T_s[0];
        float u1 = opt_s[t * 3 + 1] - T_s[1];
        float u2 = opt_s[t * 3 + 2] - T_s[2];
        float o0 = R_s[0] * u0 + R_s[3] * u1 + R_s[6] * u2;   // R^T
        float o1 = R_s[1] * u0 + R_s[4] * u1 + R_s[7] * u2;
        float o2 = R_s[2] * u0 + R_s[5] * u1 + R_s[8] * u2;
        float nm = sqrtf(o0 * o0 + o1 * o1 + o2 * o2 + 1e-8f);
        _Float16* cp = cat + (size_t)bn * CAT_;
        cp[192 + t] = (_Float16)o0;
        cp[288 + t] = (_Float16)o1;
        cp[384 + t] = (_Float16)o2;
        cp[480 + t] = (_Float16)nm;
    }
    // o_pair = a @ zn -> cat[576:2112)
    for (int t = tid; t < H_ * CZ_; t += 256) {
        int h = t >> 7, c = t & 127;
        float acc = 0.f;
        for (int k = 0; k < K_; ++k) acc += a_s[h * 64 + k] * zn_s[k * CZ_ + c];
        cat[(size_t)bn * CAT_ + 576 + t] = (_Float16)acc;
    }
}

// ---------------------------------------------------------------------------
// Kernel 5: output GEMM  (2048 x 2112 f16) @ (2112 x 384) + bout -> d_out f32
// 32x64 output block per wave (2 M-tiles x 4 N-tiles).
// ---------------------------------------------------------------------------
__global__ __launch_bounds__(32) void k_out(
    const _Float16* __restrict__ cat, const float* __restrict__ Wout,
    const float* __restrict__ bout, float* __restrict__ out)
{
    const int NG = 6;                        // 24 N-tiles / 4
    int mg = blockIdx.x / NG;                // 64 M-groups
    int ng = blockIdx.x % NG;
    int lane = threadIdx.x;
    int l16  = lane & 15;
    int aoff = (lane < 16) ? 0 : 8;
    int boff = (lane < 16) ? 0 : 16;

    const _Float16* arow0 = cat + (size_t)(mg * 32 + l16) * CAT_;
    const _Float16* arow1 = arow0 + (size_t)16 * CAT_;
    const float* wrow[4]; int col[4];
#pragma unroll
    for (int j = 0; j < 4; ++j) {
        col[j]  = (ng * 4 + j) * 16 + l16;
        wrow[j] = Wout + (size_t)col[j] * CAT_;
    }

    v8f acc0[4] = {}; v8f acc1[4] = {};
    for (int kb = 0; kb < CAT_; kb += 32) {
        v16h A0 = load_a_f16(arow0, kb, aoff);
        v16h A1 = load_a_f16(arow1, kb, aoff);
#pragma unroll
        for (int j = 0; j < 4; ++j) {
            v16h Bm = load_b_f32(wrow[j], kb, boff);
            acc0[j] = __builtin_amdgcn_wmma_f32_16x16x32_f16(false, A0, false, Bm,
                                                             (short)0, acc0[j], false, false);
            acc1[j] = __builtin_amdgcn_wmma_f32_16x16x32_f16(false, A1, false, Bm,
                                                             (short)0, acc1[j], false, false);
        }
    }

    int mb0 = mg * 32 + ((lane < 16) ? 0 : 8);
#pragma unroll
    for (int j = 0; j < 4; ++j) {
        float bv = bout[col[j]];
#pragma unroll
        for (int r = 0; r < 8; ++r) {
            out[(size_t)(mb0 + r) * CS_ + col[j]]      = acc0[j][r] + bv;
            out[(size_t)(mb0 + 16 + r) * CS_ + col[j]] = acc1[j][r] + bv;
        }
    }
}

// ---------------------------------------------------------------------------
extern "C" void kernel_launch(void* const* d_in, const int* in_sizes, int n_in,
                              void* d_out, int out_size, void* d_ws, size_t ws_size,
                              hipStream_t stream) {
    (void)in_sizes; (void)n_in; (void)out_size; (void)ws_size;

    const float* s     = (const float*)d_in[0];
    const float* z     = (const float*)d_in[1];
    const float* rot   = (const float*)d_in[2];
    const float* trans = (const float*)d_in[3];
    const float* mask  = (const float*)d_in[4];
    // d_in[5] = rel_pos (unused)
    const float* Wq    = (const float*)d_in[6];
    const float* bq    = (const float*)d_in[7];
    const float* Wkv   = (const float*)d_in[8];
    const float* bkv   = (const float*)d_in[9];
    const float* Wqp   = (const float*)d_in[10];
    const float* bqp   = (const float*)d_in[11];
    const float* Wkvp  = (const float*)d_in[12];
    const float* bkvp  = (const float*)d_in[13];
    const float* Wb    = (const float*)d_in[14];
    const float* bb    = (const float*)d_in[15];
    const float* hwv   = (const float*)d_in[16];
    const float* Wout  = (const float*)d_in[17];
    const float* bout  = (const float*)d_in[18];
    float* out = (float*)d_out;

    char* base = (char*)d_ws;
    size_t off = 0;
    auto alloc = [&](size_t bytes) -> void* {
        void* p = base + off;
        off = (off + bytes + 255) & ~(size_t)255;
        return p;
    };
    float*     qb    = (float*)alloc((size_t)BN_ * 192 * 4);
    float*     kvb   = (float*)alloc((size_t)BN_ * 384 * 4);
    float*     qpb   = (float*)alloc((size_t)BN_ * 144 * 4);
    float*     kvpb  = (float*)alloc((size_t)BN_ * 432 * 4);
    float*     q_pts = (float*)alloc((size_t)BN_ * 144 * 4);
    float*     k_pts = (float*)alloc((size_t)BN_ * 144 * 4);
    float*     v_pts = (float*)alloc((size_t)BN_ * 288 * 4);
    int*       nidx  = (int*)  alloc((size_t)BN_ * K_  * 4);
    _Float16*  cat   = (_Float16*)alloc((size_t)BN_ * CAT_ * 2);

    // 1) fused projections (WMMA, 32x64 block per wave)
    k_proj<<<dim3(64 * 18), dim3(32), 0, stream>>>(
        s, Wq, bq, Wkv, bkv, Wqp, bqp, Wkvp, bkvp, qb, kvb, qpb, kvpb);

    // 2) rigid point transforms
    k_points<<<dim3((BN_ * 192) / 256), dim3(256), 0, stream>>>(
        qpb, kvpb, rot, trans, q_pts, k_pts, v_pts);

    // 3) top-K neighbors
    k_topk<<<dim3(BN_), dim3(256), 0, stream>>>(trans, nidx);

    // 4) gathered attention -> f16 concat rows
    k_attn<<<dim3(BN_), dim3(256), 0, stream>>>(
        qb, kvb, q_pts, k_pts, v_pts, z, rot, trans, mask, nidx, Wb, bb, hwv, cat);

    // 5) output GEMM (WMMA, 32x64 block per wave) + bias -> d_out
    k_out<<<dim3(64 * 6), dim3(32), 0, stream>>>(cat, Wout, bout, out);
}
